// Net2_22651657519481
// MI455X (gfx1250) — matrix operand order repaired and checked by reading, not compiled, not run on hardware
//
#include <hip/hip_runtime.h>
#include <hip/hip_bf16.h>

// ---------------------------------------------------------------------------
// Sizes (fixed by the reference problem)
// ---------------------------------------------------------------------------
#define NN 50000
#define EE 800000
#define FF 128
#define CC 40
#define LL 3

typedef __bf16 bf16_t;
typedef bf16_t v16bf __attribute__((ext_vector_type(16)));
typedef float  v8f   __attribute__((ext_vector_type(8)));

union ABfrag { v16bf v; uint4 q[2]; };

__device__ __forceinline__ unsigned short f2bf(float x) {
    unsigned int u = __float_as_uint(x);
    unsigned int r = u + 0x7FFFu + ((u >> 16) & 1u);   // round-to-nearest-even
    return (unsigned short)(r >> 16);
}

// ---------------------------------------------------------------------------
// Degree count: deg[dst] += 1  (deg buffer pre-zeroed with hipMemsetAsync)
// ---------------------------------------------------------------------------
__global__ void k_deg(const int* __restrict__ dst, float* __restrict__ deg, int E) {
    int e = blockIdx.x * blockDim.x + threadIdx.x;
    if (e < E) atomicAdd(&deg[dst[e]], 1.0f);
}

// dis = (deg+1)^-1/2 (in place over deg buffer), selfn = 1/(deg+1)
__global__ void k_disself(float* __restrict__ dis, float* __restrict__ selfn, int n) {
    int i = blockIdx.x * blockDim.x + threadIdx.x;
    if (i < n) {
        float d = dis[i] + 1.0f;
        dis[i]   = rsqrtf(d);
        selfn[i] = 1.0f / d;
    }
}

// norm[e] = dis[src[e]] * dis[dst[e]]
__global__ void k_norm(const int* __restrict__ src, const int* __restrict__ dst,
                       const float* __restrict__ dis, float* __restrict__ norm, int E) {
    int e = blockIdx.x * blockDim.x + threadIdx.x;
    if (e < E) norm[e] = dis[src[e]] * dis[dst[e]];
}

// new_w0[i][j] = relu(w0[i] * E_meta[j] + b0[i][j])   (128x128)
__global__ void k_neww0(const float* __restrict__ w0, const float* __restrict__ emeta,
                        const float* __restrict__ b0, float* __restrict__ nw0) {
    int idx = blockIdx.x * blockDim.x + threadIdx.x;   // 16384 threads
    int i = idx >> 7, j = idx & 127;
    float v = w0[i] * emeta[j] + b0[idx];
    nw0[idx] = v > 0.0f ? v : 0.0f;
}

// ---------------------------------------------------------------------------
// Pack a row-major fp32 weight matrix W[128][128] into WMMA bf16 B-fragment
// order. Flat dword index: ct*1024 + kc*256 + lane*8 + v.
//   lane<16: column n=lane,    K half = 0..15  -> K = kc*32 + 2v (+1)
//   lane>=16: column n=lane-16, K half = 16..31 -> K = kc*32 + 16 + 2v (+1)
// ---------------------------------------------------------------------------
__global__ void k_packB(const float* __restrict__ W, unsigned int* __restrict__ pk) {
    int idx = blockIdx.x * blockDim.x + threadIdx.x;   // 8192 threads
    int v    =  idx        & 7;
    int lane = (idx >> 3)  & 31;
    int kc   = (idx >> 8)  & 3;
    int ct   = (idx >> 10) & 7;
    int n     = lane & 15;
    int khalf = (lane >> 4) * 16;
    int k0    = kc * 32 + khalf + 2 * v;
    int col   = ct * 16 + n;
    unsigned int lo = f2bf(W[k0 * 128 + col]);
    unsigned int hi = f2bf(W[(k0 + 1) * 128 + col]);
    pk[idx] = lo | (hi << 16);
}

// fp32 -> bf16 elementwise
__global__ void k_f2bf(const float* __restrict__ in, unsigned short* __restrict__ out, int n) {
    int i = blockIdx.x * blockDim.x + threadIdx.x;
    if (i < n) out[i] = f2bf(in[i]);
}

// relu(agg) -> bf16 (A matrix for next layer)
__global__ void k_relubf(const float* __restrict__ agg, unsigned short* __restrict__ out, int n) {
    int i = blockIdx.x * blockDim.x + threadIdx.x;
    if (i < n) {
        float v = agg[i];
        out[i] = f2bf(v > 0.0f ? v : 0.0f);
    }
}

// ---------------------------------------------------------------------------
// WMMA GEMM: out[M,128] = A[M,128] @ W[128,128]  with fused epilogues.
//   mode 0: outbf = bf16(result)                        (h = x @ new_w0)
//   mode 1: hw = result + bias; agg = hw * selfn[row]   (GCN layer transform)
// Block: 256 threads = 8 waves; wave w owns column tile [16w, 16w+16);
// block owns 16 rows. K = 128 = 4 chunks of 32 (v_wmma_f32_16x16x32_bf16).
// ---------------------------------------------------------------------------
__global__ __launch_bounds__(256) void k_gemm_wmma(
    const unsigned short* __restrict__ Abf,   // [M,128] bf16 row-major
    const unsigned int*  __restrict__ Bpk,    // 8192 packed dwords
    const float* __restrict__ bias,           // [128]   (mode 1)
    const float* __restrict__ selfn,          // [M]     (mode 1)
    float* __restrict__ hw,                   // [M,128] (mode 1)
    float* __restrict__ agg,                  // [M,128] (mode 1)
    unsigned short* __restrict__ outbf,       // [M,128] (mode 0)
    int mode)
{
    __shared__ __align__(16) unsigned short Atile[16 * 128];   // 4 KB

    const int tid  = threadIdx.x;
    const int lane = tid & 31;
    const int wv   = tid >> 5;                 // column tile 0..7
    const long rowBase = (long)blockIdx.x * 16;

    // Stage the 16x128 bf16 A tile: 256 threads x 16B = 4096B, fully coalesced.
    {
        const uint4* g = reinterpret_cast<const uint4*>(Abf + rowBase * 128);
        uint4* s = reinterpret_cast<uint4*>(Atile);
        s[tid] = g[tid];
    }
    __syncthreads();

    const int arow = lane & 15;
    const int aoff = (lane >> 4) * 16;         // K sub-offset for this lane half

    v8f c = {};
    #pragma unroll
    for (int kc = 0; kc < 4; ++kc) {
        ABfrag a, b;
        // A fragment: two contiguous 8-bf16 runs from LDS (2x ds_load_b128)
        const uint4* ap = reinterpret_cast<const uint4*>(&Atile[arow * 128 + kc * 32 + aoff]);
        a.q[0] = ap[0];
    a.q[1] = ap[1];                        // +8 elements = +1 uint4
        // B fragment: 8 contiguous dwords per lane from packed weights
        const uint4* bp = reinterpret_cast<const uint4*>(Bpk + ((wv * 4 + kc) * 256 + lane * 8));
        b.q[0] = bp[0];
        b.q[1] = bp[1];
        c = __builtin_amdgcn_wmma_f32_16x16x32_bf16(
                false, a.v, false, b.v, (short)0, c, false, false);
    }

    // C/D layout: lane L, vgpr r -> M = r + (L>=16)*8, N = L&15
    const int colBase = wv * 16;
    const int n   = lane & 15;
    const int mof = (lane >> 4) * 8;

    if (mode == 0) {
        #pragma unroll
        for (int r = 0; r < 8; ++r) {
            long node = rowBase + mof + r;
            outbf[node * 128 + colBase + n] = f2bf(c[r]);
        }
    } else {
        float bv = bias[colBase + n];
        #pragma unroll
        for (int r = 0; r < 8; ++r) {
            long node = rowBase + mof + r;
            float v = c[r] + bv;
            hw[node * 128 + colBase + n]  = v;
            agg[node * 128 + colBase + n] = v * selfn[node];
        }
    }
}

// ---------------------------------------------------------------------------
// Edge scatter: agg[dst] += hw[src] * norm[e].
// One wave per edge, lane l handles features [4l, 4l+4). norm/src/dst loads
// are wave-uniform -> coalesced; hw read is a 16B vector load; atomics are
// no-return global_atomic_add_f32 (L2-resident working set).
// ---------------------------------------------------------------------------
__global__ __launch_bounds__(256) void k_scatter(
    const int* __restrict__ src, const int* __restrict__ dst,
    const float* __restrict__ norm, const float* __restrict__ hw,
    float* __restrict__ agg, int E)
{
    long t = (long)blockIdx.x * blockDim.x + threadIdx.x;
    int e = (int)(t >> 5);
    if (e >= E) return;
    int f = (int)(t & 31) * 4;
    float nrm = norm[e];
    long s = src[e], d = dst[e];
    const float4 v = *reinterpret_cast<const float4*>(hw + s * 128 + f);
    float* a = agg + d * 128 + f;
    atomicAdd(a + 0, v.x * nrm);
    atomicAdd(a + 1, v.y * nrm);
    atomicAdd(a + 2, v.z * nrm);
    atomicAdd(a + 3, v.w * nrm);
}

// ---------------------------------------------------------------------------
// Final: logits = relu(agg) @ lt1_W + lt1_b, then log_softmax over 40 classes.
// One wave per node (8 nodes / 256-thread block); h row cached in LDS.
// ---------------------------------------------------------------------------
__global__ __launch_bounds__(256) void k_final(
    const float* __restrict__ agg, const float* __restrict__ W,
    const float* __restrict__ b, float* __restrict__ out)
{
    __shared__ float hrow[8][128];
    __shared__ float lg[8][CC];
    __shared__ float lse[8];
    const int lane = threadIdx.x & 31;
    const int wv   = threadIdx.x >> 5;
    const long node = (long)blockIdx.x * 8 + wv;     // 50000 = 6250*8 exact

    #pragma unroll
    for (int k = 0; k < 4; ++k) {
        float v = agg[node * 128 + lane * 4 + k];
        hrow[wv][lane * 4 + k] = v > 0.0f ? v : 0.0f;
    }
    __syncthreads();

    for (int c = lane; c < CC; c += 32) {
        float acc = b[c];
        #pragma unroll 8
        for (int k = 0; k < 128; ++k) acc += hrow[wv][k] * W[k * CC + c];
        lg[wv][c] = acc;
    }
    __syncthreads();

    if (lane == 0) {
        float m = -INFINITY;
        for (int c = 0; c < CC; ++c) m = fmaxf(m, lg[wv][c]);
        float s = 0.0f;
        for (int c = 0; c < CC; ++c) s += __expf(lg[wv][c] - m);
        lse[wv] = m + __logf(s);
    }
    __syncthreads();

    for (int c = lane; c < CC; c += 32)
        out[node * CC + c] = lg[wv][c] - lse[wv];
}

// ---------------------------------------------------------------------------
// Host launcher
// ---------------------------------------------------------------------------
extern "C" void kernel_launch(void* const* d_in, const int* in_sizes, int n_in,
                              void* d_out, int out_size, void* d_ws, size_t ws_size,
                              hipStream_t stream) {
    const float* x      = (const float*)d_in[0];            // [N,128]
    const int*   eidx   = (const int*)  d_in[1];            // [2,E]
    const float* emeta  = (const float*)d_in[2];            // [128]
    const float* w0     = (const float*)d_in[3];            // [128]
    const float* b0     = (const float*)d_in[4];            // [128,128]
    const float* gcnW   = (const float*)d_in[5];            // [3,128,128]
    const float* gcnb   = (const float*)d_in[6];            // [3,128]
    const float* lt1W   = (const float*)d_in[7];            // [128,40]
    const float* lt1b   = (const float*)d_in[8];            // [40]
    float* out = (float*)d_out;

    const int* src = eidx;
    const int* dst = eidx + EE;

    // ---- workspace carve-up (256B aligned) ----
    char* ws = (char*)d_ws;
    size_t off = 0;
    auto take = [&](size_t bytes) {
        void* p = ws + off;
        off = (off + bytes + 255) & ~size_t(255);
        return p;
    };
    float*          dis   = (float*)take(NN * 4);                 // deg -> dis
    float*          selfn = (float*)take(NN * 4);
    float*          norm  = (float*)take(EE * 4);
    float*          nw0   = (float*)take(FF * FF * 4);
    unsigned int*   Bpk   = (unsigned int*)take(4 * 8192 * 4);    // 4 packed mats
    unsigned short* xbf   = (unsigned short*)take((size_t)NN * 128 * 2);
    unsigned short* hbf   = (unsigned short*)take((size_t)NN * 128 * 2);
    float*          hw    = (float*)take((size_t)NN * 128 * 4);
    float*          agg   = (float*)take((size_t)NN * 128 * 4);
    (void)ws_size; (void)n_in; (void)in_sizes; (void)out_size;

    // ---- graph normalization ----
    hipMemsetAsync(dis, 0, NN * 4, stream);
    k_deg    <<<(EE + 255) / 256, 256, 0, stream>>>(dst, dis, EE);
    k_disself<<<(NN + 255) / 256, 256, 0, stream>>>(dis, selfn, NN);
    k_norm   <<<(EE + 255) / 256, 256, 0, stream>>>(src, dst, dis, norm, EE);

    // ---- dynamic weights + weight packing (bf16 B fragments) ----
    k_neww0<<<64, 256, 0, stream>>>(w0, emeta, b0, nw0);
    k_packB<<<32, 256, 0, stream>>>(nw0, Bpk);
    for (int l = 0; l < LL; ++l)
        k_packB<<<32, 256, 0, stream>>>(gcnW + (size_t)l * FF * FF, Bpk + (1 + l) * 8192);

    // ---- h = x @ new_w0 via WMMA ----
    k_f2bf<<<(NN * 128 + 255) / 256, 256, 0, stream>>>(x, xbf, NN * 128);
    k_gemm_wmma<<<NN / 16, 256, 0, stream>>>(xbf, Bpk, nullptr, nullptr,
                                             nullptr, nullptr, hbf, 0);

    // ---- GCN layers ----
    const int scatterGrid = (int)(((long)EE * 32 + 255) / 256);
    for (int l = 0; l < LL; ++l) {
        k_gemm_wmma<<<NN / 16, 256, 0, stream>>>(hbf, Bpk + (1 + l) * 8192,
                                                 gcnb + l * FF, selfn,
                                                 hw, agg, nullptr, 1);
        k_scatter<<<scatterGrid, 256, 0, stream>>>(src, dst, norm, hw, agg, EE);
        if (l < LL - 1)
            k_relubf<<<(NN * 128 + 255) / 256, 256, 0, stream>>>(agg, hbf, NN * 128);
    }

    // ---- classifier + log_softmax (relu fused into load) ----
    k_final<<<NN / 8, 256, 0, stream>>>(agg, lt1W, lt1b, out);
}
